// GnnModel_81423989998122
// MI455X (gfx1250) — matrix operand order repaired and checked by reading, not compiled, not run on hardware
//
#include <hip/hip_runtime.h>
#include <cstddef>
#include <cstdint>

// ---------------------------------------------------------------------------
// MEGNet-style GNN forward on MI455X (gfx1250, wave32, WMMA).
// BF16 WMMA GEMMs with FP32 accumulation. Weights pre-swizzled to the exact
// per-lane WMMA B-operand layout (kt-major => one contiguous 32KB chunk per
// K-step), then streamed into an LDS double buffer with CDNA5 async-to-LDS
// DMA (global_load_async_to_lds_b128 / s_wait_asynccnt) so WMMA never waits
// on L2 latency. Fused 2-layer MLP per workgroup: X(LDS) -> H(LDS) -> out.
// Workspace required: ~128.5 MB.
// ---------------------------------------------------------------------------

typedef __attribute__((ext_vector_type(16))) __bf16 v16bf;
typedef __attribute__((ext_vector_type(8)))  __bf16 v8bf;
typedef __attribute__((ext_vector_type(8)))  float  v8f;

#define DEV __device__ __forceinline__

constexpr int NB   = 16;    // graphs
constexpr int NS   = 512;   // sites per graph
constexpr int NE   = 6144;  // edges per graph
constexpr int DIM  = 256;   // embed dim D
constexpr int NHID = 512;   // 2D hidden
constexpr int NRBF = 128;
constexpr int NMSG = 6;
constexpr int ROWS = 64;    // edge-rows per workgroup in fused kernels

// -------- workspace layout (bytes) --------
constexpr size_t OFF_BOND  = 0;                               // [B*E, D] f32
constexpr size_t SZ_BOND   = (size_t)NB * NE * DIM * 4;
constexpr size_t OFF_ATOM  = OFF_BOND + SZ_BOND;              // [B*N, D] f32
constexpr size_t SZ_ATOM   = (size_t)NB * NS * DIM * 4;
constexpr size_t OFF_DELTA = OFF_ATOM + SZ_ATOM;              // [B*N, D] f32
constexpr size_t OFF_WBOND = OFF_DELTA + SZ_ATOM;             // bf16 swizzled
constexpr size_t SZ_WBOND  = (size_t)NRBF * DIM * 2;
constexpr size_t OFF_WE1   = OFF_WBOND + SZ_WBOND;
constexpr size_t SZ_WE1    = (size_t)NMSG * 3 * DIM * NHID * 2;
constexpr size_t OFF_WE2   = OFF_WE1 + SZ_WE1;
constexpr size_t SZ_WE2    = (size_t)NMSG * NHID * DIM * 2;
constexpr size_t OFF_WN1   = OFF_WE2 + SZ_WE2;
constexpr size_t SZ_WN1    = (size_t)NMSG * NHID * NHID * 2;
constexpr size_t OFF_WN2   = OFF_WN1 + SZ_WN1;

// ---------------------------------------------------------------------------
// CDNA5 async global->LDS DMA (ASYNCcnt-tracked). Each lane moves 16B.
DEV void async_chunk(const __bf16* __restrict__ g, const __bf16* l,
                     int tid, int units /* 16B units, multiple of 256 */) {
  const unsigned lbase = (unsigned)(uintptr_t)l;   // LDS offset = addr[31:0]
  for (int u = tid; u < units; u += 256) {
    asm volatile("global_load_async_to_lds_b128 %0, %1, off"
                 :: "v"(lbase + (unsigned)u * 16), "v"(g + (size_t)u * 8)
                 : "memory");
  }
}
DEV void wait_async0() { asm volatile("s_wait_asynccnt 0" ::: "memory"); }

// ---------------------------------------------------------------------------
// Convert 8 contiguous f32 -> 8 bf16 and store (16B LDS store).
DEV void cvt8(const float* __restrict__ src, __bf16* dst) {
  float4 f0 = *(const float4*)(src);
  float4 f1 = *(const float4*)(src + 4);
  v8bf o;
  o[0] = (__bf16)f0.x; o[1] = (__bf16)f0.y; o[2] = (__bf16)f0.z; o[3] = (__bf16)f0.w;
  o[4] = (__bf16)f1.x; o[5] = (__bf16)f1.y; o[6] = (__bf16)f1.z; o[7] = (__bf16)f1.w;
  *(v8bf*)dst = o;
}

// ---------------------------------------------------------------------------
// One K-step (K=32) of WMMA against an LDS-resident weight chunk.
// A-fragment (16-bit 16x32): lane l holds row m=l&15, K=(l>>4)*8+j (j<8) and
// K=16+(l>>4)*8+(j-8). B-fragment: lane l holds the same K as A, N=l&15 ->
// 16 contiguous bf16 per lane in the swizzled chunk.
template<int CT>
DEV void gemm_step(const __bf16* xk, const __bf16* Wch, int ntb, int lane,
                   v8f (&acc)[CT]) {
  v8bf alo = *(const v8bf*)(xk);
  v8bf ahi = *(const v8bf*)(xk + 16);
  v16bf a;
#pragma unroll
  for (int j = 0; j < 8; ++j) { a[j] = alo[j]; a[j + 8] = ahi[j]; }
  const __bf16* wk = Wch + ((size_t)ntb * 32 + lane) * 16;
#pragma unroll
  for (int t = 0; t < CT; ++t) {
    v16bf b = *(const v16bf*)(wk + (size_t)t * 512);
    acc[t] = __builtin_amdgcn_wmma_f32_16x16x32_bf16(
        false, a, false, b, (short)0, acc[t], false, false);
  }
}

// ---------------------------------------------------------------------------
// Full GEMM: acc += X[64 x KDIM](LDS) @ W(global, swizzled, NTtot col-tiles),
// weights streamed through an LDS double buffer via async DMA.
template<int KDIM, int CT, int NTtot>
DEV void gemm_pipelined(const __bf16* Xlds, const __bf16* __restrict__ Wg,
                        __bf16* Wbuf, int ntb, int mbase, int tid,
                        v8f (&acc)[CT]) {
  constexpr int KT = KDIM / 32;
  constexpr int CHUNK = NTtot * 512;   // bf16 elems per K-step chunk
  constexpr int UNITS = NTtot * 64;    // 16B units per chunk
  const int lane = tid & 31;
  const int m  = mbase + (lane & 15);
  const int kh = (lane >> 4) * 8;
  const __bf16* xrow = Xlds + (size_t)m * KDIM + kh;

  async_chunk(Wg, Wbuf, tid, UNITS);           // prologue: chunk 0
  wait_async0();
  __syncthreads();
  for (int kt = 0; kt < KT; ++kt) {
    const __bf16* cur = Wbuf + (size_t)(kt & 1) * CHUNK;
    if (kt + 1 < KT)                            // prefetch next chunk
      async_chunk(Wg + (size_t)(kt + 1) * CHUNK,
                  Wbuf + (size_t)((kt + 1) & 1) * CHUNK, tid, UNITS);
    gemm_step<CT>(xrow + kt * 32, cur, ntb, lane, acc);
    wait_async0();                              // own async issues done
    __syncthreads();                            // whole WG chunk ready/consumed
  }
}

// ---------------------------------------------------------------------------
// Swizzle f32 weight matrices [mats][K][N] -> per-lane packed bf16 WMMA layout.
__global__ __launch_bounds__(256) void swz_w(const float* __restrict__ W,
                                             __bf16* __restrict__ out,
                                             int K, int N, int total) {
  const int i = blockIdx.x * 256 + threadIdx.x;
  if (i >= total) return;
  const int NT = N >> 4, KT = K >> 5;
  const int lane = i & 31;
  int tmp = i >> 5;
  const int nt = tmp % NT; tmp /= NT;
  const int kt = tmp % KT;
  const int mat = tmp / KT;
  const float* Wm = W + (size_t)mat * K * N;
  const int n  = nt * 16 + (lane & 15);
  const int kb = kt * 32 + (lane >> 4) * 8;
  v16bf o;
#pragma unroll
  for (int j = 0; j < 16; ++j) o[j] = (__bf16)Wm[(size_t)(kb + (j < 8 ? j : j + 8)) * N + n];
  *(v16bf*)(out + (size_t)i * 16) = o;
}

// ---------------------------------------------------------------------------
// atom_state[b,n,:] = atom_emb[site_class[b,n], :]
__global__ __launch_bounds__(256) void embed_k(const int* __restrict__ cls,
                                               const float* __restrict__ emb,
                                               float* __restrict__ atom) {
  const int site = blockIdx.x * 4 + (threadIdx.x >> 6);
  const int t = (threadIdx.x & 63) * 4;
  const int c = cls[site];
  *(float4*)(atom + (size_t)site * DIM + t) =
      *(const float4*)(emb + (size_t)c * DIM + t);
}

// ---------------------------------------------------------------------------
// RBF expansion + bond_state = rbf @ bond_W + bond_b (K=128 WMMA GEMM).
// Whole 64KB weight matrix async-loaded to LDS once.
__global__ __launch_bounds__(256) void bond_init(
    const float* __restrict__ dist, const float* __restrict__ centers,
    const float* __restrict__ gap, const __bf16* __restrict__ Wswz,
    const float* __restrict__ bias, float* __restrict__ bond) {
  extern __shared__ __align__(32) char smem[];
  __bf16* R  = (__bf16*)smem;                           // [64][128] bf16
  __bf16* WB = (__bf16*)(smem + ROWS * NRBF * 2);       // [128*256] bf16
  const int tid = threadIdx.x;
  const int rowbase = blockIdx.x * ROWS;

  async_chunk(Wswz, WB, tid, NRBF * DIM / 8);           // 4096 units = 64KB

  const float g = gap[0];
  for (int i = tid; i < ROWS * NRBF; i += 256) {
    const int r = i >> 7, c = i & (NRBF - 1);
    float d = dist[rowbase + r];
    d = (d != 0.f) ? d : 1.f;
    const float x = d - centers[c];
    R[i] = (__bf16)__expf(-x * x * g);
  }
  wait_async0();
  __syncthreads();

  const int lane = tid & 31, w = tid >> 5;
  const int mt = (w & 3) * 16, ntb = (w >> 2) * 8;
  const v8f z8 = {0.f,0.f,0.f,0.f,0.f,0.f,0.f,0.f};
  v8f acc[8];
#pragma unroll
  for (int t = 0; t < 8; ++t) acc[t] = z8;
  const int m  = mt + (lane & 15);
  const int kh = (lane >> 4) * 8;
#pragma unroll
  for (int kt = 0; kt < NRBF / 32; ++kt)
    gemm_step<8>(R + (size_t)m * NRBF + kh + kt * 32,
                 WB + (size_t)kt * (DIM / 16) * 512, ntb, lane, acc);
#pragma unroll
  for (int t = 0; t < 8; ++t) {
    const int n = (ntb + t) * 16 + (lane & 15);
    const float b = bias[n];
#pragma unroll
    for (int r = 0; r < 8; ++r) {
      const int mm = mt + r + (lane >> 4) * 8;
      bond[(size_t)(rowbase + mm) * DIM + n] = acc[t][r] + b;
    }
  }
}

// ---------------------------------------------------------------------------
// Fused 2-layer MLP over 64 edge-rows per workgroup.
// EDGE=true : X = [bond | atom[idx1] | atom[idx0]] (K1=768); masked += into
//             bond_state.
// EDGE=false: X = [atom[idx1] | bond] (K1=512); atomic segment-sum into delta.
template<int K1, bool EDGE>
__global__ __launch_bounds__(256) void fused_mlp(
    const float* __restrict__ bond, const float* __restrict__ atom,
    const int* __restrict__ conn, const float* __restrict__ dist,
    const __bf16* __restrict__ W1, const float* __restrict__ B1,
    const __bf16* __restrict__ W2, const float* __restrict__ B2,
    float* __restrict__ bond_out, float* __restrict__ delta) {
  extern __shared__ __align__(32) char smem[];
  __bf16* X    = (__bf16*)smem;                                 // [64][K1]
  __bf16* H    = (__bf16*)(smem + (size_t)ROWS * K1 * 2);       // [64][512]
  __bf16* Wbuf = (__bf16*)(smem + (size_t)ROWS * (K1 + NHID) * 2); // 2x32KB
  const int tid = threadIdx.x;
  const int rowbase = blockIdx.x * ROWS;

  // ---- stage concatenated input tile to LDS as bf16 ----
  for (int i = tid; i < ROWS * (K1 / 8); i += 256) {
    const int r  = i / (K1 / 8);
    const int c8 = (i % (K1 / 8)) * 8;
    const int rg = rowbase + r;
    const int bG = rg / NE;
    const float* src;
    if (EDGE) {
      if (c8 < DIM)
        src = bond + (size_t)rg * DIM + c8;
      else if (c8 < 2 * DIM)
        src = atom + ((size_t)bG * NS + conn[rg * 2 + 1]) * DIM + (c8 - DIM);
      else
        src = atom + ((size_t)bG * NS + conn[rg * 2 + 0]) * DIM + (c8 - 2 * DIM);
    } else {
      if (c8 < DIM)
        src = atom + ((size_t)bG * NS + conn[rg * 2 + 1]) * DIM + c8;
      else
        src = bond + (size_t)rg * DIM + (c8 - DIM);
    }
    cvt8(src, X + (size_t)r * K1 + c8);
  }
  __syncthreads();

  const int lane = tid & 31;
  const int w = tid >> 5;
  const int mt = (w & 3) * 16;
  const v8f z8 = {0.f,0.f,0.f,0.f,0.f,0.f,0.f,0.f};

  // ---- phase 1: H = relu(X @ W1 + B1), H -> LDS bf16 ----
  {
    const int ntb = (w >> 2) * 16;     // wave covers 16 of 32 col-tiles
    v8f acc[16];
#pragma unroll
    for (int t = 0; t < 16; ++t) acc[t] = z8;
    gemm_pipelined<K1, 16, NHID / 16>(X, W1, Wbuf, ntb, mt, tid, acc);
#pragma unroll
    for (int t = 0; t < 16; ++t) {
      const int n = (ntb + t) * 16 + (lane & 15);
      const float b = B1[n];
#pragma unroll
      for (int r = 0; r < 8; ++r) {
        const int m = mt + r + (lane >> 4) * 8;
        float v = acc[t][r] + b;
        v = v > 0.f ? v : 0.f;
        H[(size_t)m * NHID + n] = (__bf16)v;
      }
    }
  }
  __syncthreads();

  // ---- phase 2: out = H @ W2 + B2 ----
  {
    const int ntb = (w >> 2) * 8;      // wave covers 8 of 16 col-tiles
    v8f acc[8];
#pragma unroll
    for (int t = 0; t < 8; ++t) acc[t] = z8;
    gemm_pipelined<NHID, 8, DIM / 16>(H, W2, Wbuf, ntb, mt, tid, acc);
#pragma unroll
    for (int t = 0; t < 8; ++t) {
      const int n = (ntb + t) * 16 + (lane & 15);
      const float b = B2[n];
#pragma unroll
      for (int r = 0; r < 8; ++r) {
        const int m = mt + r + (lane >> 4) * 8;
        const int rg = rowbase + m;
        const float v = acc[t][r] + b;
        if (EDGE) {
          if (dist[rg] != 0.f) bond_out[(size_t)rg * DIM + n] += v;
        } else {
          const int bG = rg / NE;
          const int t0 = conn[rg * 2 + 0];
          __hip_atomic_fetch_add(delta + ((size_t)bG * NS + t0) * DIM + n, v,
                                 __ATOMIC_RELAXED, __HIP_MEMORY_SCOPE_AGENT);
        }
      }
    }
  }
}

// ---------------------------------------------------------------------------
__global__ __launch_bounds__(256) void zero_k(float4* p, int n) {
  const int i = blockIdx.x * 256 + threadIdx.x;
  if (i < n) p[i] = make_float4(0.f, 0.f, 0.f, 0.f);
}

// atom_state += atom_mask ? delta : 0 ; delta = 0
__global__ __launch_bounds__(256) void apply_k(const int* __restrict__ cls,
                                               float* __restrict__ atom,
                                               float* __restrict__ delta) {
  const int i = blockIdx.x * 256 + threadIdx.x;  // over B*N*D/4
  const int site = i >> 6;                       // D/4 = 64
  float4 d = ((float4*)delta)[i];
  ((float4*)delta)[i] = make_float4(0.f, 0.f, 0.f, 0.f);
  if (cls[site] != 0) {
    float4 a = ((float4*)atom)[i];
    a.x += d.x; a.y += d.y; a.z += d.z; a.w += d.w;
    ((float4*)atom)[i] = a;
  }
}

// atom_mean = mean_emb[cls] + atom_state @ offset_W + offset_b; masked mean.
__global__ __launch_bounds__(256) void pool_k(
    const int* __restrict__ cls, const float* __restrict__ atom,
    const float* __restrict__ mean_emb, const float* __restrict__ offW,
    const float* __restrict__ offb, float* __restrict__ out) {
  __shared__ float ssum[256], scnt[256];
  const int b = blockIdx.x, tid = threadIdx.x;
  float sum = 0.f, cnt = 0.f;
  for (int n = tid; n < NS; n += 256) {
    const int c = cls[b * NS + n];
    if (c != 0) {
      const float* row = atom + ((size_t)b * NS + n) * DIM;
      float dot = 0.f;
      for (int d = 0; d < DIM; d += 4) {
        float4 a  = *(const float4*)(row + d);
        float4 wv = *(const float4*)(offW + d);
        dot += a.x * wv.x + a.y * wv.y + a.z * wv.z + a.w * wv.w;
      }
      sum += mean_emb[c] + dot + offb[0];
      cnt += 1.f;
    }
  }
  ssum[tid] = sum; scnt[tid] = cnt;
  __syncthreads();
  for (int s = 128; s > 0; s >>= 1) {
    if (tid < s) { ssum[tid] += ssum[tid + s]; scnt[tid] += scnt[tid + s]; }
    __syncthreads();
  }
  if (tid == 0) out[b] = ssum[0] / scnt[0];
}

// ---------------------------------------------------------------------------
extern "C" void kernel_launch(void* const* d_in, const int* in_sizes, int n_in,
                              void* d_out, int out_size, void* d_ws, size_t ws_size,
                              hipStream_t stream) {
  (void)in_sizes; (void)n_in; (void)out_size; (void)ws_size;
  const int*   site_class = (const int*)  d_in[0];
  const float* distance   = (const float*)d_in[1];
  const int*   conn       = (const int*)  d_in[2];
  const float* atom_emb   = (const float*)d_in[3];
  const float* mean_emb   = (const float*)d_in[4];
  const float* centers    = (const float*)d_in[5];
  const float* gap        = (const float*)d_in[6];
  const float* bond_W     = (const float*)d_in[7];
  const float* bond_b     = (const float*)d_in[8];
  const float* offset_W   = (const float*)d_in[9];
  const float* offset_b   = (const float*)d_in[10];
  const float* edge_W1    = (const float*)d_in[11];
  const float* edge_b1    = (const float*)d_in[12];
  const float* edge_W2    = (const float*)d_in[13];
  const float* edge_b2    = (const float*)d_in[14];
  const float* node_W1    = (const float*)d_in[15];
  const float* node_b1    = (const float*)d_in[16];
  const float* node_W2    = (const float*)d_in[17];
  const float* node_b2    = (const float*)d_in[18];
  float* out = (float*)d_out;

  char* ws = (char*)d_ws;
  float*  bond  = (float*)(ws + OFF_BOND);
  float*  atom  = (float*)(ws + OFF_ATOM);
  float*  delta = (float*)(ws + OFF_DELTA);
  __bf16* wbond = (__bf16*)(ws + OFF_WBOND);
  __bf16* we1   = (__bf16*)(ws + OFF_WE1);
  __bf16* we2   = (__bf16*)(ws + OFF_WE2);
  __bf16* wn1   = (__bf16*)(ws + OFF_WN1);
  __bf16* wn2   = (__bf16*)(ws + OFF_WN2);

  // Pre-swizzle all weights to bf16 WMMA layout (L2-resident thereafter).
  swz_w<<<2048 / 256,   256, 0, stream>>>(bond_W,  wbond, NRBF,  DIM,  2048);
  swz_w<<<147456 / 256, 256, 0, stream>>>(edge_W1, we1,   3*DIM, NHID, 147456);
  swz_w<<<49152 / 256,  256, 0, stream>>>(edge_W2, we2,   NHID,  DIM,  49152);
  swz_w<<<98304 / 256,  256, 0, stream>>>(node_W1, wn1,   NHID,  NHID, 98304);
  swz_w<<<49152 / 256,  256, 0, stream>>>(node_W2, wn2,   NHID,  DIM,  49152);

  embed_k<<<NB * NS / 4, 256, 0, stream>>>(site_class, atom_emb, atom);
  bond_init<<<(NB * NE) / ROWS, 256, ROWS * NRBF * 2 + NRBF * DIM * 2, stream>>>(
      distance, centers, gap, wbond, bond_b, bond);
  zero_k<<<(NB * NS * DIM / 4) / 256, 256, 0, stream>>>((float4*)delta,
                                                        NB * NS * DIM / 4);

  const int grid = (NB * NE) / ROWS;  // 1536 workgroups
  const int sh_edge = ROWS * (3 * DIM + NHID) * 2 + 2 * NHID * 32 * 2; // 224KB
  const int sh_node = ROWS * (2 * DIM + NHID) * 2 + 2 * NHID * 32 * 2; // 192KB
  for (int i = 0; i < NMSG; ++i) {
    fused_mlp<3 * DIM, true><<<grid, 256, sh_edge, stream>>>(
        bond, atom, conn, distance,
        we1 + (size_t)i * 3 * DIM * NHID, edge_b1 + i * NHID,
        we2 + (size_t)i * NHID * DIM,     edge_b2 + i * DIM,
        bond, delta);
    fused_mlp<2 * DIM, false><<<grid, 256, sh_node, stream>>>(
        bond, atom, conn, distance,
        wn1 + (size_t)i * NHID * NHID,    node_b1 + i * NHID,
        wn2 + (size_t)i * NHID * DIM,     node_b2 + i * DIM,
        bond, delta);
    apply_k<<<(NB * NS * DIM / 4) / 256, 256, 0, stream>>>(site_class, atom, delta);
  }
  pool_k<<<NB, 256, 0, stream>>>(site_class, atom, mean_emb, offset_W, offset_b, out);
}
